// Interaction_Block_51522427683092
// MI455X (gfx1250) — matrix operand order repaired and checked by reading, not compiled, not run on hardware
//
#include <hip/hip_runtime.h>
#include <hip/hip_bf16.h>
#include <stdint.h>

typedef unsigned int   u32;
typedef unsigned short u16;
typedef __attribute__((ext_vector_type(16))) __bf16 v16bf;
typedef __attribute__((ext_vector_type(8)))  float  v8f;
typedef __attribute__((ext_vector_type(4)))  u32    u32x4;

#define DEVFN static __device__ __forceinline__
#define AST 136   // row stride (u16) for 128-wide activation panels: 272B, bank-spreading, 16B aligned
#define ASF 40    // row stride (u16) for 32-wide feature panels

union FragU { v16bf v; u32x4 q[2]; };

struct MLP3 {
  const u16 *w0, *w1, *w2;          // pre-swizzled bf16 B-fragments
  const float *b0, *b1, *b2;        // f32 biases (read through cache)
  const float *g, *be;              // layernorm gamma/beta (optional)
};

DEVFN u16 f2bf(float x) {                       // f32 -> bf16 round-to-nearest-even
  u32 u = __float_as_uint(x);
  return (u16)((u + 0x7FFFu + ((u >> 16) & 1u)) >> 16);
}

DEVFN v8f wmma_bf16(v16bf a, v16bf b, v8f c) {
  return __builtin_amdgcn_wmma_f32_16x16x32_bf16(false, a, false, b, (short)0, c, false, false);
}

// A fragment (16x32 bf16) from row-major LDS panel. Lane l: row m=l&15, half h -> K = 8*(l>>4)+h (+8 for h>=8)
DEVFN v16bf load_a(const u16* act, int AS, int kc, int lane) {
  int m = lane & 15, g = lane >> 4;
  const u16* p = act + m * AS + kc * 32 + 8 * g;
  FragU u;
  u.q[0] = *(const u32x4*)(p);        // halves 0..7  : K = 8g .. 8g+7
  u.q[1] = *(const u32x4*)(p + 16);   // halves 8..15 : K = 16+8g .. 16+8g+7
  return u.v;
}

// B fragment (32x16 bf16) from pre-swizzled global weights: 32 contiguous bytes per lane
DEVFN v16bf load_b(const u16* wf, int fragIdx, int lane) {
  const u16* p = wf + ((size_t)fragIdx * 32 + lane) * 16;
  FragU u;
  u.q[0] = *(const u32x4*)(p);
  u.q[1] = *(const u32x4*)(p + 8);
  return u.v;
}

DEVFN void init_bias(v8f acc[8], const float* bias, int lane) {
  int n = lane & 15;
#pragma unroll
  for (int nt = 0; nt < 8; ++nt) {
    float b = bias[nt * 16 + n];
#pragma unroll
    for (int r = 0; r < 8; ++r) acc[nt][r] = b;
  }
}

DEVFN void relu8(v8f acc[8]) {
#pragma unroll
  for (int nt = 0; nt < 8; ++nt)
#pragma unroll
    for (int r = 0; r < 8; ++r) acc[nt][r] = fmaxf(acc[nt][r], 0.f);
}

// full 128-wide layer: acc += act(Kp) @ W  (8 N-tiles)
DEVFN void gemm8(const u16* act, int AS, int Kp, const u16* wf, int kcBase, v8f acc[8], int lane) {
  int KC = Kp >> 5;
  for (int kc = 0; kc < KC; ++kc) {
    v16bf a = load_a(act, AS, kc, lane);
    if (kc + 1 < KC)
      __builtin_prefetch((const void*)(wf + ((size_t)(kcBase + kc + 1) * 8 * 32) * 16), 0, 1);
#pragma unroll
    for (int nt = 0; nt < 8; ++nt) {
      v16bf b = load_b(wf, (kcBase + kc) * 8 + nt, lane);
      acc[nt] = wmma_bf16(a, b, acc[nt]);
    }
  }
}

// narrow head layer (single padded N-tile)
DEVFN void gemm1(const u16* act, int AS, int Kp, const u16* wf, v8f& acc, int lane) {
  int KC = Kp >> 5;
  for (int kc = 0; kc < KC; ++kc) {
    v16bf a = load_a(act, AS, kc, lane);
    v16bf b = load_b(wf, kc, lane);
    acc = wmma_bf16(a, b, acc);
  }
}

// layernorm over 128 cols of each of the 16 rows; rows live in one 16-lane half (vgpr r -> row r+8g)
DEVFN void ln_acc(v8f acc[8], const float* gam, const float* bet, int lane) {
  int n = lane & 15;
#pragma unroll
  for (int r = 0; r < 8; ++r) {
    float s = 0.f, s2 = 0.f;
#pragma unroll
    for (int nt = 0; nt < 8; ++nt) { float x = acc[nt][r]; s += x; s2 += x * x; }
#pragma unroll
    for (int off = 1; off < 16; off <<= 1) { s += __shfl_xor(s, off, 32); s2 += __shfl_xor(s2, off, 32); }
    float mu = s * (1.f / 128.f);
    float var = s2 * (1.f / 128.f) - mu * mu;
    float rs = rsqrtf(var + 1e-5f);
#pragma unroll
    for (int nt = 0; nt < 8; ++nt) acc[nt][r] = (acc[nt][r] - mu) * rs;
  }
#pragma unroll
  for (int nt = 0; nt < 8; ++nt) {
    float gg = gam[nt * 16 + n], bb = bet[nt * 16 + n];
#pragma unroll
    for (int r = 0; r < 8; ++r) acc[nt][r] = acc[nt][r] * gg + bb;
  }
}

DEVFN void store_act(u16* dst, int AS, const v8f acc[8], int lane) {
  int n = lane & 15, g = lane >> 4;
#pragma unroll
  for (int nt = 0; nt < 8; ++nt)
#pragma unroll
    for (int r = 0; r < 8; ++r)
      dst[(r + 8 * g) * AS + nt * 16 + n] = f2bf(acc[nt][r]);
}

DEVFN void two_hidden(const u16* in, u16* t0, u16* t1, const MLP3& M, int lane) {
  v8f acc[8];
  init_bias(acc, M.b0, lane);
  gemm8(in, AST, 128, M.w0, 0, acc, lane);
  relu8(acc);
  store_act(t0, AST, acc, lane);
  __syncthreads();
  init_bias(acc, M.b1, lane);
  gemm8(t0, AST, 128, M.w1, 0, acc, lane);
  relu8(acc);
  store_act(t1, AST, acc, lane);
  __syncthreads();
}

// ---------------------------------------------------------------- weight swizzle
__global__ void prep_frag(const float* __restrict__ W, u16* __restrict__ dst,
                          int K, int N, int Kp, int Np) {
  int i = blockIdx.x * 256 + threadIdx.x;
  int tot = Kp * Np;
  if (i >= tot) return;
  int h = i & 15;
  int lane = (i >> 4) & 31;
  int fr = i >> 9;
  int NT = Np >> 4;
  int kc = fr / NT, nt = fr - kc * NT;
  int k  = kc * 32 + ((h >> 3) << 4) + ((lane >> 4) << 3) + (h & 7);   // K = 8*(lane>=16) + h (+8 for h>=8) within chunk
  int nn = nt * 16 + (lane & 15);
  float v = (k < K && nn < N) ? W[(size_t)k * N + nn] : 0.f;
  dst[i] = f2bf(v);
}

// ---------------------------------------------------------------- node-level kernel
struct NodeArgs {
  const float* nodeLat;
  MLP3 nw, mi, ii, dv;
  float *wnode, *minv, *iinv, *dvext;
  int N;
};

DEVFN void head_scalar(const u16* in, u16* t0, u16* t1, const MLP3& M,
                       float* out, int base, int N, int lane, bool sig) {
  two_hidden(in, t0, t1, M, lane);
  int n = lane & 15, g = lane >> 4;
  v8f a;
  float bb = (n < 1) ? M.b2[0] : 0.f;
#pragma unroll
  for (int r = 0; r < 8; ++r) a[r] = bb;
  gemm1(t1, AST, 128, M.w2, a, lane);
  if (n == 0) {
#pragma unroll
    for (int r = 0; r < 8; ++r) {
      int nd = base + r + 8 * g;
      if (nd < N) {
        float v = a[r];
        if (sig) v = 1.f / (1.f + __expf(-v));
        out[nd] = v;
      }
    }
  }
  __syncthreads();
}

DEVFN void head_vec3(const u16* in, u16* t0, u16* t1, const MLP3& M,
                     float* out, int base, int N, int lane) {
  two_hidden(in, t0, t1, M, lane);
  int n = lane & 15, g = lane >> 4;
  v8f a;
  float bb = (n < 3) ? M.b2[n] : 0.f;
#pragma unroll
  for (int r = 0; r < 8; ++r) a[r] = bb;
  gemm1(t1, AST, 128, M.w2, a, lane);
  if (n < 3) {
#pragma unroll
    for (int r = 0; r < 8; ++r) {
      int nd = base + r + 8 * g;
      if (nd < N) out[(size_t)nd * 3 + n] = a[r];
    }
  }
  __syncthreads();
}

__global__ __launch_bounds__(64) void node_kernel(NodeArgs A) {
  __shared__ u16 s_act[2][16 * AST], s_t0[2][16 * AST], s_t1[2][16 * AST];
  int wave = threadIdx.x >> 5, lane = threadIdx.x & 31;
  int m = lane & 15, g = lane >> 4;
  int base = (blockIdx.x * 2 + wave) * 16;
  u16 *act = s_act[wave], *t0 = s_t0[wave], *t1 = s_t1[wave];
  {
    int node = base + m; if (node >= A.N) node = A.N - 1;
    const float* p = A.nodeLat + (size_t)node * 128;
    for (int q = 0; q < 64; ++q) { int c = g * 64 + q; act[m * AST + c] = f2bf(p[c]); }
  }
  __syncthreads();
  head_scalar(act, t0, t1, A.nw, A.wnode, base, A.N, lane, true);
  head_scalar(act, t0, t1, A.mi, A.minv,  base, A.N, lane, false);
  head_scalar(act, t0, t1, A.ii, A.iinv,  base, A.N, lane, false);
  head_vec3  (act, t0, t1, A.dv, A.dvext, base, A.N, lane);
}

// ---------------------------------------------------------------- fused edge kernel
struct EdgeArgs {
  const float *spos, *rpos, *edx, *eattr;
  const float *va, *vb, *vc;
  const float *sv[6];
  const float *rv[6];
  const float *nodeLat;
  const int   *eidx;
  MLP3 ee, ef, ie, d1, d2, fs;
  const float *wnode;
  float *accF, *accT, *outIL;
  int E;
};

DEVFN void head_edge(const u16* il, u16* t0, u16* t1, const MLP3& M,
                     float* scr, int Nr, int lane) {
  two_hidden(il, t0, t1, M, lane);
  int n = lane & 15, g = lane >> 4;
  v8f a;
  float bb = (n < Nr) ? M.b2[n] : 0.f;
#pragma unroll
  for (int r = 0; r < 8; ++r) a[r] = bb;
  gemm1(t1, AST, 128, M.w2, a, lane);
#pragma unroll
  for (int r = 0; r < 8; ++r) scr[(r + 8 * g) * 16 + n] = a[r];
  __syncthreads();
}

__global__ __launch_bounds__(64) void fused_edge_kernel(EdgeArgs A) {
  __shared__ u16 s_t0[2][16 * AST], s_t1[2][16 * AST], s_il[2][16 * AST];
  __shared__ u16 s_ft[2][3 * 16 * ASF];
  int wave = threadIdx.x >> 5, lane = threadIdx.x & 31;
  int m = lane & 15, g = lane >> 4, n = m;
  int e0 = (blockIdx.x * 2 + wave) * 16;
  int e = e0 + m; if (e >= A.E) e = A.E - 1;
  u16 *t0 = s_t0[wave], *t1 = s_t1[wave], *il = s_il[wave];
  u16 *featS = s_ft[wave], *featR = featS + 16 * ASF, *featE = featR + 16 * ASF;

  // basis vectors (kept in registers; reused by physics stage)
  float ax = A.va[3 * (size_t)e], ay = A.va[3 * (size_t)e + 1], az = A.va[3 * (size_t)e + 2];
  float bx = A.vb[3 * (size_t)e], by = A.vb[3 * (size_t)e + 1], bz = A.vb[3 * (size_t)e + 2];
  float cx = A.vc[3 * (size_t)e], cy = A.vc[3 * (size_t)e + 1], cz = A.vc[3 * (size_t)e + 2];

  // ---- features: half-wave g=0 -> sender side, g=1 -> receiver side (signed)
  {
    u16* frow = (g ? featR : featS) + m * ASF;
#pragma unroll
    for (int j = 0; j < 6; ++j) {
      const float* vp = (g ? A.rv[j] : A.sv[j]) + 3 * (size_t)e;
      float vx = vp[0], vy = vp[1], vz = vp[2];
      float sg = (g && j != 3 && j != 5) ? -1.f : 1.f;   // receivers: -,-,-,+,-,+
      frow[3 * j + 0] = f2bf(sg * (ax * vx + ay * vy + az * vz));
      frow[3 * j + 1] = f2bf(sg * (bx * vx + by * vy + bz * vz));
      frow[3 * j + 2] = f2bf(sg * (cx * vx + cy * vy + cz * vz));
    }
    for (int k = 18; k < 32; ++k) frow[k] = 0;
    if (!g) {
      u16* er = featE + m * ASF;
      float dx = A.edx[3 * (size_t)e], dy = A.edx[3 * (size_t)e + 1], dz = A.edx[3 * (size_t)e + 2];
      er[0] = f2bf(sqrtf(dx * dx + dy * dy + dz * dz));
#pragma unroll
      for (int t = 0; t < 4; ++t) er[1 + t] = f2bf(A.eattr[4 * (size_t)e + t]);
      for (int k = 5; k < 32; ++k) er[k] = 0;
    }
  }
  __syncthreads();

  v8f accI[8];                        // interaction layer-1 accumulator (384-K split in 3 panels)
  init_bias(accI, A.ie.b0, lane);
  v8f acc[8];

  // ---- edge_encoder chain  ->  feeds panel C (rows 256..383) of interaction W0
  init_bias(acc, A.ee.b0, lane); gemm8(featE, ASF, 32, A.ee.w0, 0, acc, lane); relu8(acc);
  store_act(t0, AST, acc, lane); __syncthreads();
  init_bias(acc, A.ee.b1, lane); gemm8(t0, AST, 128, A.ee.w1, 0, acc, lane); relu8(acc);
  store_act(t1, AST, acc, lane); __syncthreads();
  init_bias(acc, A.ee.b2, lane); gemm8(t1, AST, 128, A.ee.w2, 0, acc, lane);
  ln_acc(acc, A.ee.g, A.ee.be, lane);
  store_act(t0, AST, acc, lane); __syncthreads();
  gemm8(t0, AST, 128, A.ie.w0, 8, accI, lane);
  __syncthreads();

  // ---- node_latent[s] + node_latent[r]  ->  panel B (rows 128..255)
  {
    int s = A.eidx[e], r = A.eidx[A.E + e];
    const float* ps = A.nodeLat + (size_t)s * 128;
    const float* pr = A.nodeLat + (size_t)r * 128;
    for (int q = 0; q < 64; ++q) { int c = g * 64 + q; t0[m * AST + c] = f2bf(ps[c] + pr[c]); }
  }
  __syncthreads();
  gemm8(t0, AST, 128, A.ie.w0, 4, accI, lane);
  __syncthreads();

  // ---- sender latent (edge_feat_encoder, LN) -> stash
  init_bias(acc, A.ef.b0, lane); gemm8(featS, ASF, 32, A.ef.w0, 0, acc, lane); relu8(acc);
  store_act(t0, AST, acc, lane); __syncthreads();
  init_bias(acc, A.ef.b1, lane); gemm8(t0, AST, 128, A.ef.w1, 0, acc, lane); relu8(acc);
  store_act(t1, AST, acc, lane); __syncthreads();
  init_bias(acc, A.ef.b2, lane); gemm8(t1, AST, 128, A.ef.w2, 0, acc, lane);
  ln_acc(acc, A.ef.g, A.ef.be, lane);
  v8f stash[8];
#pragma unroll
  for (int nt = 0; nt < 8; ++nt) stash[nt] = acc[nt];

  // ---- receiver latent, sum with stash  ->  panel A (rows 0..127)
  init_bias(acc, A.ef.b0, lane); gemm8(featR, ASF, 32, A.ef.w0, 0, acc, lane); relu8(acc);
  store_act(t0, AST, acc, lane); __syncthreads();
  init_bias(acc, A.ef.b1, lane); gemm8(t0, AST, 128, A.ef.w1, 0, acc, lane); relu8(acc);
  store_act(t1, AST, acc, lane); __syncthreads();
  init_bias(acc, A.ef.b2, lane); gemm8(t1, AST, 128, A.ef.w2, 0, acc, lane);
  ln_acc(acc, A.ef.g, A.ef.be, lane);
#pragma unroll
  for (int nt = 0; nt < 8; ++nt) acc[nt] += stash[nt];
  store_act(t0, AST, acc, lane); __syncthreads();
  gemm8(t0, AST, 128, A.ie.w0, 0, accI, lane);

  // ---- interaction MLP tail
  relu8(accI);
  store_act(t0, AST, accI, lane); __syncthreads();
  init_bias(acc, A.ie.b1, lane); gemm8(t0, AST, 128, A.ie.w1, 0, acc, lane); relu8(acc);
  store_act(t1, AST, acc, lane); __syncthreads();
  init_bias(acc, A.ie.b2, lane); gemm8(t1, AST, 128, A.ie.w2, 0, acc, lane);
  ln_acc(acc, A.ie.g, A.ie.be, lane);
  // interaction latent: f32 to output, bf16 stays resident for the decoder heads
#pragma unroll
  for (int nt = 0; nt < 8; ++nt)
#pragma unroll
    for (int r = 0; r < 8; ++r) {
      int row = e0 + r + 8 * g;
      if (row < A.E) A.outIL[(size_t)row * 128 + nt * 16 + n] = acc[nt][r];
    }
  store_act(il, AST, acc, lane); __syncthreads();

  // ---- decoder heads (feature LDS region reused as f32 scratch)
  float* scr = (float*)featS;                  // 3 * 16x16 f32 = 3072B <= 3840B
  head_edge(il, t0, t1, A.d1, scr + 0,   3, lane);
  head_edge(il, t0, t1, A.d2, scr + 256, 3, lane);
  head_edge(il, t0, t1, A.fs, scr + 512, 1, lane);

  // ---- per-edge physics + scatter (lanes 0..15, one edge each)
  if (!g && (e0 + m) < A.E) {
    const float* c1 = scr + m * 16;
    const float* c2 = scr + 256 + m * 16;
    float lam = scr[512 + m * 16];
    float fx = c1[0] * ax + c1[1] * bx + c1[2] * cx;
    float fy = c1[0] * ay + c1[1] * by + c1[2] * cy;
    float fz = c1[0] * az + c1[1] * bz + c1[2] * cz;
    float qx = c2[0] * ax + c2[1] * bx + c2[2] * cx;
    float qy = c2[0] * ay + c2[1] * by + c2[2] * cy;
    float qz = c2[0] * az + c2[1] * bz + c2[2] * cz;
    int s = A.eidx[e], r = A.eidx[A.E + e];
    float ws = A.wnode[s], wr = A.wnode[r];
    float spx = A.spos[3 * (size_t)e], spy = A.spos[3 * (size_t)e + 1], spz = A.spos[3 * (size_t)e + 2];
    float rpx = A.rpos[3 * (size_t)e], rpy = A.rpos[3 * (size_t)e + 1], rpz = A.rpos[3 * (size_t)e + 2];
    float inv = 1.f / (ws + wr + 1e-8f);
    float lx = rpx - (ws * spx + wr * rpx) * inv;
    float ly = rpy - (ws * spy + wr * rpy) * inv;
    float lz = rpz - (ws * spz + wr * rpz) * inv;
    float gx = fx * lam, gy = fy * lam, gz = fz * lam;
    float tqx = ly * gz - lz * gy;
    float tqy = lz * gx - lx * gz;
    float tqz = lx * gy - ly * gx;
    atomicAdd(&A.accF[3 * (size_t)r + 0], fx);
    atomicAdd(&A.accF[3 * (size_t)r + 1], fy);
    atomicAdd(&A.accF[3 * (size_t)r + 2], fz);
    atomicAdd(&A.accT[3 * (size_t)r + 0], qx - tqx);
    atomicAdd(&A.accT[3 * (size_t)r + 1], qy - tqy);
    atomicAdd(&A.accT[3 * (size_t)r + 2], qz - tqz);
  }
}

// ---------------------------------------------------------------- final combine
__global__ void combine_kernel(const float* minv, const float* iinv, const float* dvext,
                               const float* accF, const float* accT,
                               float* outDV, float* outDW, int N) {
  int i = blockIdx.x * 256 + threadIdx.x;
  if (i >= N * 3) return;
  int nd = i / 3;
  outDV[i] = minv[nd] * accF[i] + dvext[i];
  outDW[i] = iinv[nd] * accT[i];
}

// ---------------------------------------------------------------- launch
extern "C" void kernel_launch(void* const* d_in, const int* in_sizes, int n_in,
                              void* d_out, int out_size, void* d_ws, size_t ws_size,
                              hipStream_t stream) {
  (void)n_in; (void)out_size; (void)ws_size;
  const int E = in_sizes[0] / 3;      // 800000
  const int N = in_sizes[19] / 128;   // 50000
  auto F = [&](int i) { return (const float*)d_in[i]; };

  char* ws = (char*)d_ws;
  size_t off = 0;
  auto alloc = [&](size_t bytes) -> void* {
    off = (off + 255) & ~(size_t)255;
    void* p = ws + off; off += bytes; return p;
  };
  auto prep = [&](int widx, int K, int Nn) -> const u16* {
    int Kp = (K + 31) & ~31, Np = (Nn + 15) & ~15;
    u16* dst = (u16*)alloc((size_t)Kp * Np * sizeof(u16));
    int tot = Kp * Np;
    prep_frag<<<(tot + 255) / 256, 256, 0, stream>>>(F(widx), dst, K, Nn, Kp, Np);
    return dst;
  };
  // params flattened in sorted-key order at index 21: per MLP -> W0,b0,W1,b1,W2,b2[,gamma,beta]
  auto mk = [&](int base, int K0, int NL, bool ln) -> MLP3 {
    MLP3 m;
    m.w0 = prep(base + 0, K0, 128);  m.b0 = F(base + 1);
    m.w1 = prep(base + 2, 128, 128); m.b1 = F(base + 3);
    m.w2 = prep(base + 4, 128, NL);  m.b2 = F(base + 5);
    m.g  = ln ? F(base + 6) : nullptr;
    m.be = ln ? F(base + 7) : nullptr;
    return m;
  };

  MLP3 dv = mk(21, 128, 3, false);    // dv_ext_decoder
  MLP3 ee = mk(27, 5,   128, true);   // edge_encoder
  MLP3 ef = mk(35, 18,  128, true);   // edge_feat_encoder
  MLP3 fs = mk(43, 128, 1, false);    // f_scaler
  MLP3 d1 = mk(49, 128, 3, false);    // i1_decoder
  MLP3 d2 = mk(55, 128, 3, false);    // i2_decoder
  MLP3 ii = mk(61, 128, 1, false);    // i_inv_decoder
  MLP3 ie = mk(67, 384, 128, true);   // interaction_encoder
  MLP3 mi = mk(75, 128, 1, false);    // m_inv_decoder
  MLP3 nw = mk(81, 128, 1, false);    // node_weight_decoder

  float* wnode = (float*)alloc((size_t)N * 4);
  float* minv  = (float*)alloc((size_t)N * 4);
  float* iinv  = (float*)alloc((size_t)N * 4);
  float* dvext = (float*)alloc((size_t)N * 3 * 4);
  float* accF  = (float*)alloc((size_t)N * 3 * 4);
  float* accT  = (float*)alloc((size_t)N * 3 * 4);

  hipMemsetAsync(accF, 0, (size_t)N * 3 * 4, stream);
  hipMemsetAsync(accT, 0, (size_t)N * 3 * 4, stream);

  NodeArgs na;
  na.nodeLat = F(19);
  na.nw = nw; na.mi = mi; na.ii = ii; na.dv = dv;
  na.wnode = wnode; na.minv = minv; na.iinv = iinv; na.dvext = dvext; na.N = N;
  node_kernel<<<(N + 31) / 32, 64, 0, stream>>>(na);

  EdgeArgs ea;
  ea.spos = F(0); ea.rpos = F(1); ea.edx = F(2); ea.eattr = F(18);
  ea.va = F(3); ea.vb = F(4); ea.vc = F(5);
  for (int j = 0; j < 6; ++j) { ea.sv[j] = F(6 + j); ea.rv[j] = F(12 + j); }
  ea.nodeLat = F(19);
  ea.eidx = (const int*)d_in[20];
  ea.ee = ee; ea.ef = ef; ea.ie = ie; ea.d1 = d1; ea.d2 = d2; ea.fs = fs;
  ea.wnode = wnode; ea.accF = accF; ea.accT = accT;
  float* outDV = (float*)d_out;
  float* outDW = outDV + (size_t)N * 3;
  ea.outIL = outDV + (size_t)N * 6;
  ea.E = E;
  fused_edge_kernel<<<(E + 31) / 32, 64, 0, stream>>>(ea);

  combine_kernel<<<(N * 3 + 255) / 256, 256, 0, stream>>>(minv, iinv, dvext, accF, accT, outDV, outDW, N);
}
